// Attend_40922448396897
// MI455X (gfx1250) — compile-verified, matrix operand order checked by reading, not run on hardware
//
#include <hip/hip_runtime.h>
#include <hip/hip_bf16.h>

// Problem constants (reference: B,H,N,D = 2,16,2048,64)
constexpr int B = 2;
constexpr int H = 16;
constexpr int N = 2048;
constexpr int D = 64;
constexpr int NT = N / 16;                 // 128 row-tiles per (b,h)
constexpr int ND = N * D;                  // flattened (n,d) per (b,h)
constexpr float SCALE = 0.03125f;          // 1/(sqrt(16)*sqrt(64)) = 1/32

typedef __attribute__((ext_vector_type(16))) __bf16 v16bf;
typedef __attribute__((ext_vector_type(8)))  __bf16 v8bf;
typedef __attribute__((ext_vector_type(8)))  float  v8f;

#define WMMA_BF16(A, Bm, C) \
    __builtin_amdgcn_wmma_f32_16x16x32_bf16(false, (A), false, (Bm), (short)0, (C), false, false)

#if __has_builtin(__builtin_amdgcn_global_load_async_to_lds_b128) && \
    __has_builtin(__builtin_amdgcn_s_wait_asynccnt)
#define USE_ASYNC_V 1
// builtin signature (from hipcc diagnostic): (int4 AS1*, int4 AS3*, imm off, imm cpol)
typedef int v4i_ld __attribute__((__vector_size__(4 * sizeof(int))));
typedef __attribute__((address_space(1))) v4i_ld* gptr_v4i;
typedef __attribute__((address_space(3))) v4i_ld* lptr_v4i;
#else
#define USE_ASYNC_V 0
#endif

__device__ inline float fast_rcp(float x) {
#if __has_builtin(__builtin_amdgcn_rcpf)
    return __builtin_amdgcn_rcpf(x);
#else
    return 1.f / x;
#endif
}

// ---------------------------------------------------------------------------
// Pass 0a via WMMA: q' = (scale*W_pre) q ; k' = W_pre k over the head axis.
// One wave per 16-column chunk of the flattened (n,d) axis:
//   D(16e x 16x) = W(16x16, K zero-padded to 32) @ Q(16f x 16x)
// ---------------------------------------------------------------------------
__global__ __launch_bounds__(256) void premix_qk_kernel(
    const float* __restrict__ q, const float* __restrict__ k,
    const float* __restrict__ Wpre,
    __bf16* __restrict__ qb, __bf16* __restrict__ kb)
{
    const int wave = threadIdx.x >> 5;
    const int lane = threadIdx.x & 31;
    const int half = lane >> 4;
    const int l16  = lane & 15;

    int c = blockIdx.x * 8 + wave;           // chunk id in [0, B*ND/16)
    int b  = c / (ND / 16);
    int x0 = (c % (ND / 16)) * 16;

    // A-fragments of W_pre (rows=e, K=f<16 real, f>=16 zero); scale folded for q
    v16bf awq, awk;
#pragma unroll
    for (int t = 0; t < 16; ++t) { awq[t] = (__bf16)0.f; awk[t] = (__bf16)0.f; }
#pragma unroll
    for (int t = 0; t < 8; ++t) {
        float wv = Wpre[l16 * H + half * 8 + t];
        awq[t] = (__bf16)(wv * SCALE);
        awk[t] = (__bf16)wv;
    }

    // B-fragments: column x = x0+l16, K rows f per half pattern
    v16bf bq, bk;
#pragma unroll
    for (int t = 0; t < 16; ++t) { bq[t] = (__bf16)0.f; bk[t] = (__bf16)0.f; }
#pragma unroll
    for (int t = 0; t < 8; ++t) {
        int f = half * 8 + t;
        size_t off = (size_t)(b * H + f) * ND + x0 + l16;
        bq[t] = (__bf16)q[off];
        bk[t] = (__bf16)k[off];
    }

    v8f dq = {0.f, 0.f, 0.f, 0.f, 0.f, 0.f, 0.f, 0.f};
    v8f dk = {0.f, 0.f, 0.f, 0.f, 0.f, 0.f, 0.f, 0.f};
    dq = WMMA_BF16(awq, bq, dq);
    dk = WMMA_BF16(awk, bk, dk);

#pragma unroll
    for (int r = 0; r < 8; ++r) {
        int e = r + 8 * half;
        size_t off = (size_t)(b * H + e) * ND + x0 + l16;
        qb[off] = (__bf16)dq[r];
        kb[off] = (__bf16)dk[r];
    }
}

// Pass 0b: v -> bf16, transposed per head: Vt[b,h,d,j] = v[b,h,j,d]
__global__ __launch_bounds__(256) void transpose_v_kernel(
    const float* __restrict__ v, __bf16* __restrict__ vt)
{
    size_t idx = (size_t)blockIdx.x * blockDim.x + threadIdx.x;   // over B*H*D*N
    if (idx >= (size_t)B * H * D * N) return;
    int j = (int)(idx % N);
    size_t t = idx / N;
    int d = (int)(t % D);
    int bh = (int)(t / D);
    vt[idx] = (__bf16)v[((size_t)bh * N + j) * D + d];
}

// ---------------------------------------------------------------------------
// Fragment loader, row-major bf16 source with 32-element K window.
// ---------------------------------------------------------------------------
__device__ inline v16bf load_frag_row(const __bf16* __restrict__ rowptr, int half)
{
    const __bf16* p = rowptr + half * 8;
    v8bf lo = *(const v8bf*)(p);        // +0..7
    v8bf hi = *(const v8bf*)(p + 16);   // +16..23
    v16bf a;
#pragma unroll
    for (int t = 0; t < 8; ++t) { a[t] = lo[t]; a[t + 8] = hi[t]; }
    return a;
}

// ---------------------------------------------------------------------------
// Pass 1: per-(b,h,row) softmax statistics (rowmax m, 1/rowsum), flash loop.
// ---------------------------------------------------------------------------
__global__ __launch_bounds__(256) void stats_kernel(
    const __bf16* __restrict__ qb, const __bf16* __restrict__ kb,
    float* __restrict__ Mrow, float* __restrict__ Linv)
{
    const int wave = threadIdx.x >> 5;
    const int lane = threadIdx.x & 31;
    const int half = lane >> 4;
    const int l16  = lane & 15;

    int tile = blockIdx.x * 8 + wave;        // [0, B*H*NT)
    int it = tile % NT;
    int bh = tile / NT;

    const __bf16* qrow = qb + ((size_t)bh * N + (it * 16 + l16)) * D;
    v16bf aq0 = load_frag_row(qrow, half);
    v16bf aq1 = load_frag_row(qrow + 32, half);

    float m[8], l[8];
#pragma unroll
    for (int r = 0; r < 8; ++r) { m[r] = -__builtin_inff(); l[r] = 0.f; }

    for (int jt = 0; jt <= it; ++jt) {
        const __bf16* krow = kb + ((size_t)bh * N + (jt * 16 + l16)) * D;
        if (jt < it) __builtin_prefetch(krow + 16 * D, 0, 1);   // global_prefetch_b8
        v16bf bk0 = load_frag_row(krow, half);
        v16bf bk1 = load_frag_row(krow + 32, half);

        v8f s = {0.f, 0.f, 0.f, 0.f, 0.f, 0.f, 0.f, 0.f};
        s = WMMA_BF16(aq0, bk0, s);
        s = WMMA_BF16(aq1, bk1, s);

        if (jt == it) {                       // causal mask on diagonal tile
#pragma unroll
            for (int r = 0; r < 8; ++r)
                if (l16 > r + 8 * half) s[r] = -__builtin_inff();
        }
#pragma unroll
        for (int r = 0; r < 8; ++r) {
            float v = s[r];
            float tm = v;                     // row-max across the 16-lane half
            tm = fmaxf(tm, __shfl_xor(tm, 1, 32));
            tm = fmaxf(tm, __shfl_xor(tm, 2, 32));
            tm = fmaxf(tm, __shfl_xor(tm, 4, 32));
            tm = fmaxf(tm, __shfl_xor(tm, 8, 32));
            float mnew = fmaxf(m[r], tm);
            float e = __expf(v - mnew);
            e += __shfl_xor(e, 1, 32);
            e += __shfl_xor(e, 2, 32);
            e += __shfl_xor(e, 4, 32);
            e += __shfl_xor(e, 8, 32);
            l[r] = l[r] * __expf(m[r] - mnew) + e;
            m[r] = mnew;
        }
    }

    if (l16 == 0) {                           // lane 0 -> rows 0..7, lane 16 -> 8..15
#pragma unroll
        for (int r = 0; r < 8; ++r) {
            size_t row = (size_t)bh * N + it * 16 + 8 * half + r;
            Mrow[row] = m[r];
            Linv[row] = fast_rcp(l[r]);
        }
    }
}

// ---------------------------------------------------------------------------
// Pass 2: one block per (b, 16-row i-tile); 16 waves, wave w = head w.
// Per 32-wide j step, per wave: 4 WMMA (S) -> Pl -> 2 WMMA (head mix) -> Pm
// -> 4 WMMA (PV).  V fragments staged into per-wave LDS via async-to-LDS
// (ASYNCcnt) when available, so the fetch overlaps S/softmax/mix.
// ---------------------------------------------------------------------------
__global__ __launch_bounds__(512) void attend_out_kernel(
    const __bf16* __restrict__ qb, const __bf16* __restrict__ kb,
    const __bf16* __restrict__ vt, const float* __restrict__ Wpost,
    const float* __restrict__ Mrow, const float* __restrict__ Linv,
    float* __restrict__ out)
{
    __shared__ __align__(16) __bf16 Pl[512][H];    // [x=i*32+j][f]  16 KB
    __shared__ __align__(16) __bf16 Pm[H][512];    // [e][x]         16 KB
#if USE_ASYNC_V
    __shared__ __align__(16) __bf16 Vs[H][32][64]; // per-wave V stage, 64 KB
#endif

    const int it = blockIdx.x % NT;
    const int b  = blockIdx.x / NT;
    const int w  = threadIdx.x >> 5;          // head index
    const int lane = threadIdx.x & 31;
    const int half = lane >> 4;
    const int l16  = lane & 15;
    const int bh = b * H + w;

    float mr[8], li[8];
#pragma unroll
    for (int r = 0; r < 8; ++r) {
        size_t row = (size_t)bh * N + it * 16 + 8 * half + r;
        mr[r] = Mrow[row];
        li[r] = Linv[row];
    }

    // Constant A-fragment of W_post (rows=e, K=f zero-padded to 32)
    v16bf awp;
#pragma unroll
    for (int t = 0; t < 16; ++t) awp[t] = (__bf16)0.f;
#pragma unroll
    for (int t = 0; t < 8; ++t) awp[t] = (__bf16)Wpost[l16 * H + half * 8 + t];

    const __bf16* qrow = qb + ((size_t)bh * N + (it * 16 + l16)) * D;
    v16bf aq0 = load_frag_row(qrow, half);
    v16bf aq1 = load_frag_row(qrow + 32, half);

    v8f acc[4];
#pragma unroll
    for (int dc = 0; dc < 4; ++dc)
        acc[dc] = (v8f){0.f, 0.f, 0.f, 0.f, 0.f, 0.f, 0.f, 0.f};

    const int jend = it * 16;                 // start of diagonal tile
    for (int j0 = 0; j0 <= jend; j0 += 32) {
#if USE_ASYNC_V
        // kick off this step's V fragment bytes into the per-wave LDS stage:
        // lo 16B -> Vs[w][lane][dc*16], hi 16B -> Vs[w][lane][dc*16+8]
#pragma unroll
        for (int dc = 0; dc < 4; ++dc) {
            const __bf16* vrow = vt + ((size_t)bh * D + dc * 16 + l16) * N + j0 + half * 8;
            __builtin_amdgcn_global_load_async_to_lds_b128(
                (gptr_v4i)(vrow),      (lptr_v4i)(&Vs[w][lane][dc * 16]),     0, 0);
            __builtin_amdgcn_global_load_async_to_lds_b128(
                (gptr_v4i)(vrow + 16), (lptr_v4i)(&Vs[w][lane][dc * 16 + 8]), 0, 0);
        }
#endif
        // ---- scores + normalized probs for head w -> Pl[x][w] ----
#pragma unroll
        for (int s = 0; s < 2; ++s) {
            int jc = j0 + 16 * s;
            if (jc <= jend) {
                const __bf16* krow = kb + ((size_t)bh * N + (jc + l16)) * D;
                v16bf bk0 = load_frag_row(krow, half);
                v16bf bk1 = load_frag_row(krow + 32, half);
                v8f sv = {0.f, 0.f, 0.f, 0.f, 0.f, 0.f, 0.f, 0.f};
                sv = WMMA_BF16(aq0, bk0, sv);
                sv = WMMA_BF16(aq1, bk1, sv);
#pragma unroll
                for (int r = 0; r < 8; ++r) {
                    int iloc = r + 8 * half;
                    int ig = it * 16 + iloc;
                    int jg = jc + l16;
                    float p = (jg <= ig) ? __expf(sv[r] - mr[r]) * li[r] : 0.f;
                    Pl[iloc * 32 + 16 * s + l16][w] = (__bf16)p;
                }
            } else {
#pragma unroll
                for (int r = 0; r < 8; ++r)
                    Pl[(r + 8 * half) * 32 + 16 * s + l16][w] = (__bf16)0.f;
            }
        }
        if (j0 + 32 <= jend)                   // prefetch next step's K rows
            __builtin_prefetch(kb + ((size_t)bh * N + (j0 + 32 + l16)) * D, 0, 1);
        __syncthreads();

        // ---- talking-heads mix via WMMA: wave w handles chunks 2w, 2w+1 ----
#pragma unroll
        for (int cc = 0; cc < 2; ++cc) {
            int c = 2 * w + cc;               // position chunk [0,32)
            v16bf bp;
#pragma unroll
            for (int t = 0; t < 16; ++t) bp[t] = (__bf16)0.f;
            v8bf pv8 = *(const v8bf*)&Pl[c * 16 + l16][half * 8];
#pragma unroll
            for (int t = 0; t < 8; ++t) bp[t] = pv8[t];
            v8f dm = {0.f, 0.f, 0.f, 0.f, 0.f, 0.f, 0.f, 0.f};
            dm = WMMA_BF16(awp, bp, dm);
#pragma unroll
            for (int r = 0; r < 8; ++r)
                Pm[r + 8 * half][c * 16 + l16] = (__bf16)dm[r];
        }
        __syncthreads();

        // ---- PV: A-frag of P~_w from Pm (two b128 LDS loads) ----
        v16bf ap;
        {
            v8bf a0 = *(const v8bf*)&Pm[w][l16 * 32 + half * 8];
            v8bf a1 = *(const v8bf*)&Pm[w][l16 * 32 + half * 8 + 16];
#pragma unroll
            for (int t = 0; t < 8; ++t) { ap[t] = a0[t]; ap[t + 8] = a1[t]; }
        }
#if USE_ASYNC_V
        __builtin_amdgcn_s_wait_asynccnt(0);   // V stage landed in LDS
#pragma unroll
        for (int dc = 0; dc < 4; ++dc) {
            v16bf bv = *(const v16bf*)&Vs[w][lane][dc * 16];
            acc[dc] = WMMA_BF16(ap, bv, acc[dc]);
        }
#else
#pragma unroll
        for (int dc = 0; dc < 4; ++dc) {
            const __bf16* vrow = vt + ((size_t)bh * D + dc * 16 + l16) * N + j0;
            v16bf bv = load_frag_row(vrow, half);   // tail overrun <=64B lands in ws; probs are 0 there
            acc[dc] = WMMA_BF16(ap, bv, acc[dc]);
        }
#endif
        __syncthreads();
    }

#pragma unroll
    for (int dc = 0; dc < 4; ++dc)
#pragma unroll
        for (int r = 0; r < 8; ++r) {
            size_t row = (size_t)bh * N + it * 16 + r + 8 * half;
            out[row * D + dc * 16 + l16] = acc[dc][r];
        }
}

// ---------------------------------------------------------------------------
extern "C" void kernel_launch(void* const* d_in, const int* in_sizes, int n_in,
                              void* d_out, int out_size, void* d_ws, size_t ws_size,
                              hipStream_t stream)
{
    (void)in_sizes; (void)n_in; (void)out_size; (void)ws_size;
    const float* q     = (const float*)d_in[0];
    const float* k     = (const float*)d_in[1];
    const float* v     = (const float*)d_in[2];
    const float* Wpre  = (const float*)d_in[3];
    const float* Wpost = (const float*)d_in[4];
    float* out = (float*)d_out;

    const size_t qkv = (size_t)B * H * N * D;
    char* ws = (char*)d_ws;
    __bf16* qb = (__bf16*)ws;  ws += qkv * sizeof(__bf16);
    __bf16* kb = (__bf16*)ws;  ws += qkv * sizeof(__bf16);
    __bf16* vt = (__bf16*)ws;  ws += qkv * sizeof(__bf16);
    float* Mrow = (float*)ws;  ws += (size_t)B * H * N * sizeof(float);
    float* Linv = (float*)ws;

    // Pass 0: head-mix q,k via WMMA (scale folded); transpose+convert v
    premix_qk_kernel<<<B * (ND / 16) / 8, 256, 0, stream>>>(q, k, Wpre, qb, kb);
    transpose_v_kernel<<<(unsigned)((qkv + 255) / 256), 256, 0, stream>>>(v, vt);
    // Pass 1: softmax row statistics (m, 1/l) per head
    stats_kernel<<<B * H * NT / 8, 256, 0, stream>>>(qb, kb, Mrow, Linv);
    // Pass 2: mixed-prob attention output
    attend_out_kernel<<<B * NT, 512, 0, stream>>>(qb, kb, vt, Wpost, Mrow, Linv, out);
}
